// MyConv2D_86457691669130
// MI455X (gfx1250) — matrix operand order repaired and checked by reading, not compiled, not run on hardware
//
#include <hip/hip_runtime.h>

// Problem constants (from reference)
#define NB   16
#define CIN  64
#define KOUT 64
#define HW   224
#define TLW  32   // output pixels per block (along W)
#define TLH  16   // output rows per block (along H)

typedef __attribute__((ext_vector_type(16))) __bf16       v16bf;
typedef __attribute__((ext_vector_type(8)))  float        v8f;
typedef __attribute__((ext_vector_type(4)))  unsigned int u32x4;
typedef __attribute__((ext_vector_type(8)))  int          i32x8;
typedef __attribute__((ext_vector_type(4)))  int          i32x4;

// LDS layout (bytes):
//   [0)        wlds  bf16 [rs*64+k][c]  36864 el -> 73728 B
//   [73728)    xlds  bf16 4 slots x [34 px][64 c] = 8704 el -> 17408 B
//   [91136)    wscr  f32  9216 el (TDM staging scratch) -> 36864 B
#define W_ELEMS   (9 * 64 * 64)
#define X_SLOT    (34 * 64)
#define X_ELEMS   (4 * X_SLOT)
#define WSCR_OFF  ((W_ELEMS + X_ELEMS) * 2)     // 91136
#define WSCR_F32  9216                          // 16 out-ch worth of f32 weights
#define LDS_BYTES (WSCR_OFF + WSCR_F32 * 4)     // 128000 B

// XOR swizzle: 8 chunks of 8 channels per 64-ch row; kills bank conflicts on
// both element-wise staging stores and per-lane b128 fragment reads.
__device__ __forceinline__ int sw(int row, int c) {
  return row * 64 + ((((c >> 3) ^ row) & 7) << 3) + (c & 7);
}

__device__ __forceinline__ u32x4 lds128(const __bf16* base, int row, int c) {
  return *reinterpret_cast<const u32x4*>(base + sw(row, c)); // 16B aligned
}

__device__ __forceinline__ int slot4(int hh) { return (hh + 4) & 3; }

__global__ __launch_bounds__(256) void conv_wmma_bf16(
    const float* __restrict__ X, const float* __restrict__ Wg,
    const float* __restrict__ Bg, float* __restrict__ Out) {
  extern __shared__ __bf16 smem[];
  __bf16* wlds = smem;                 // swizzled bf16 weights
  __bf16* xlds = smem + W_ELEMS;       // rolling 4-row halo buffer
  float*  wscr = reinterpret_cast<float*>(reinterpret_cast<char*>(smem) + WSCR_OFF);

  const int tid  = threadIdx.x;
  const int lane = tid & 31;       // wave32
  const int l16  = lane & 15;
  const int hiH  = (lane >> 4) & 1;
  const int wave = tid >> 5;       // 8 waves
  const int mt   = wave & 3;       // 4 M-tiles (16 out-ch each)
  const int nt   = wave >> 2;      // 2 N-tiles (16 px each)

  const int w0 = blockIdx.x * TLW;
  const int h0 = blockIdx.y * TLH;
  const int n  = blockIdx.z;

  // ---- stage all weights once: TDM DMA -> LDS f32 scratch -> swizzled bf16 ----
#if __has_builtin(__builtin_amdgcn_tensor_load_to_lds) && __has_builtin(__builtin_amdgcn_s_wait_tensorcnt)
  for (int chunk = 0; chunk < 4; ++chunk) {
    if (wave == 0) {
      unsigned long long ga =
          (unsigned long long)(const void*)Wg + (unsigned long long)chunk * (WSCR_F32 * 4ull);
      // D# group0: count=1, lds_addr, 57-bit global addr, type=2
      u32x4 g0;
      g0.x = 1u;
      g0.y = (unsigned)WSCR_OFF;
      g0.z = (unsigned)(ga & 0xFFFFFFFFu);
      g0.w = (unsigned)((ga >> 32) & 0x1FFFFFFu) | (2u << 30);
      // D# group1: data_size=4B; 1-D tile: tensor_dim0=tile_dim0=9216, dim1=1
      i32x8 g1 = {
          0x20000,                                  // data_size=2 (4B), no flags
          (int)((WSCR_F32 & 0xFFFF) << 16),         // tensor_dim0[15:0] << 16
          (int)((WSCR_F32 >> 16) | (1 << 16)),      // tensor_dim0 hi | tensor_dim1 lo
          (int)((WSCR_F32 & 0xFFFF) << 16),         // tensor_dim1 hi(0) | tile_dim0
          1,                                        // tile_dim1=1, tile_dim2=0
          WSCR_F32,                                 // tensor_dim0_stride lo
          0, 0};
      i32x4 gz4 = {0, 0, 0, 0};
      i32x8 gz8 = {0, 0, 0, 0, 0, 0, 0, 0};
      __builtin_amdgcn_tensor_load_to_lds(g0, g1, gz4, gz4, gz8, 0);
      __builtin_amdgcn_s_wait_tensorcnt(0);
    }
    __syncthreads();
    for (int idx = tid; idx < WSCR_F32; idx += 256) {
      float v  = wscr[idx];
      int gidx = chunk * WSCR_F32 + idx;  // OIHW linear order
      int k    = gidx / 576;
      int rem  = gidx - k * 576;
      int c    = rem / 9;
      int rs   = rem - c * 9;
      wlds[sw(rs * 64 + k, c)] = (__bf16)v;
    }
    __syncthreads();
  }
#else
  for (int idx = tid; idx < W_ELEMS; idx += 256) {
    int k   = idx / 576;
    int rem = idx - k * 576;
    int c   = rem / 9;
    int rs  = rem - c * 9;
    wlds[sw(rs * 64 + k, c)] = (__bf16)Wg[idx];
  }
#endif

  // bias registers: acc VGPR j holds out-channel mt*16 + hiH*8 + j
  float bias[8];
#pragma unroll
  for (int j = 0; j < 8; ++j) bias[j] = Bg[mt * 16 + hiH * 8 + j];

  const float* xbase = X + (size_t)n * CIN * HW * HW;
  float*       obase = Out + (size_t)n * KOUT * HW * HW;

  // ---- prologue: stage halo rows h0-1, h0, h0+1 into slots ----
#pragma unroll
  for (int row = 0; row < 3; ++row) {
    const int hh = h0 - 1 + row;
    const int sl = slot4(hh);
    for (int idx = tid; idx < X_SLOT; idx += 256) {
      int px = idx % 34;            // coalesced along W
      int c  = idx / 34;
      int ww = w0 - 1 + px;
      float v = 0.0f;
      if ((unsigned)hh < HW && (unsigned)ww < HW)
        v = xbase[(size_t)c * (HW * HW) + hh * HW + ww];
      xlds[sw(sl * 34 + px, c)] = (__bf16)v;
    }
  }
  __syncthreads();

  for (int it = 0; it < TLH; ++it) {
    const int h = h0 + it;
    const bool doPre = (it < TLH - 1);
    const int hp = h + 2;           // prefetched row (slot == (h-2)&3, free now)

    // 1) issue prefetch global loads for row h+2 (overlaps with WMMAs below)
    float pre[9];
#pragma unroll
    for (int j = 0; j < 9; ++j) {
      int idx = tid + j * 256;
      if (doPre && idx < X_SLOT) {
        int px = idx % 34;
        int c  = idx / 34;
        int ww = w0 - 1 + px;
        float v = 0.0f;
        if ((unsigned)hp < HW && (unsigned)ww < HW)
          v = xbase[(size_t)c * (HW * HW) + hp * HW + ww];
        pre[j] = v;
      } else {
        pre[j] = 0.0f;
      }
    }

    // 2) compute: 9 taps x 2 K-chunks = 18 WMMA
    v8f acc = {};
    const int px0 = nt * 16 + l16;
#pragma unroll
    for (int r = 0; r < 3; ++r) {
      const int slr = slot4(h - 1 + r);
#pragma unroll
      for (int s = 0; s < 3; ++s) {
#pragma unroll
        for (int cb = 0; cb < 64; cb += 32) {
          union { v16bf v; u32x4 q[2]; } a, b;
          // A: 16x32 bf16; lanes 0-15: K=c+0..7 / c+16..23, lanes 16-31: +8
          const int rowA = (r * 3 + s) * 64 + mt * 16 + l16;
          const int aoff = cb + hiH * 8;
          a.q[0] = lds128(wlds, rowA, aoff);
          a.q[1] = lds128(wlds, rowA, aoff + 16);
          // B: 32x16 bf16; lanes 0-15: K=c+0..7 / c+16..23, lanes 16-31: +8
          const int rowB = slr * 34 + s + px0;
          const int boff = cb + hiH * 8;
          b.q[0] = lds128(xlds, rowB, boff);
          b.q[1] = lds128(xlds, rowB, boff + 16);
          acc = __builtin_amdgcn_wmma_f32_16x16x32_bf16(
              false, a.v, false, b.v, (short)0, acc, false, false);
        }
      }
    }

    // epilogue: C/D layout -> NCHW store, coalesced along W
    const int w = w0 + px0;
#pragma unroll
    for (int j = 0; j < 8; ++j) {
      const int k = mt * 16 + hiH * 8 + j;
      obase[(size_t)k * (HW * HW) + h * HW + w] = acc[j] + bias[j];
    }

    // 3) commit prefetched row h+2 into its slot
    if (doPre) {
      const int sl = slot4(hp);
#pragma unroll
      for (int j = 0; j < 9; ++j) {
        int idx = tid + j * 256;
        if (idx < X_SLOT) {
          int px = idx % 34;
          int c  = idx / 34;
          xlds[sw(sl * 34 + px, c)] = (__bf16)pre[j];
        }
      }
    }
    __syncthreads();
  }
}

extern "C" void kernel_launch(void* const* d_in, const int* in_sizes, int n_in,
                              void* d_out, int out_size, void* d_ws, size_t ws_size,
                              hipStream_t stream) {
  const float* x   = (const float*)d_in[0];
  const float* W   = (const float*)d_in[1];
  const float* b   = (const float*)d_in[2];
  float*       out = (float*)d_out;

  (void)in_sizes; (void)n_in; (void)out_size; (void)d_ws; (void)ws_size;

  (void)hipFuncSetAttribute((const void*)conv_wmma_bf16,
                            hipFuncAttributeMaxDynamicSharedMemorySize,
                            LDS_BYTES);

  dim3 grid(HW / TLW, HW / TLH, NB);  // 7 x 14 x 16
  conv_wmma_bf16<<<grid, 256, LDS_BYTES, stream>>>(x, W, b, out);
}
